// ASPPDeformable_23115513987485
// MI455X (gfx1250) — matrix operand order repaired and checked by reading, not compiled, not run on hardware
//
#include <hip/hip_runtime.h>
#include <hip/hip_bf16.h>
#include <math.h>

// ---------------------------------------------------------------------------
// ASPP + deformable-conv head for gfx1250 (MI455X).
// All GEMM-shaped math (offset/mask convs, deform GEMMs, 1x1 fusion conv)
// runs on v_wmma_f32_16x16x32_f16 with f32 accumulation; operands staged in LDS.
// ---------------------------------------------------------------------------

typedef __attribute__((ext_vector_type(16))) _Float16 v16h;
typedef __attribute__((ext_vector_type(8)))  float    v8f;

#define EPSBN 1e-5f
#define Bn   4
#define Cin  64
#define Hh   128
#define Ww   128
#define NPIX (Hh * Ww)      // 16384
#define Pc   256
#define CATC 1280

// ---------------- workspace layout (bytes, every segment 256B-aligned) ------
constexpr size_t SZ_CAT  = (size_t)Bn * NPIX * CATC * 2;       // 167,772,160
constexpr size_t SZ_C3   = (size_t)3 * Bn * 9 * NPIX * 4;      //   7,077,888
constexpr size_t SZ_C1   = (size_t)Bn * NPIX * 4;              //     262,144
constexpr size_t WS_CAT  = 0;
constexpr size_t WS_PY3  = WS_CAT  + SZ_CAT;
constexpr size_t WS_PX3  = WS_PY3  + SZ_C3;
constexpr size_t WS_MK3  = WS_PX3  + SZ_C3;
constexpr size_t WS_PY1  = WS_MK3  + SZ_C3;
constexpr size_t WS_PX1  = WS_PY1  + SZ_C1;
constexpr size_t WS_MK1  = WS_PX1  + SZ_C1;
constexpr size_t WS_WPK1 = WS_MK1  + SZ_C1;                    // 16*2*32*16*2   = 32768
constexpr size_t WS_WPK3 = WS_WPK1 + 32768;                    // 48*18*32*16*2  = 884736
constexpr size_t WS_WPKF = WS_WPK3 + 884736;                   // 4*40*32*16*2   = 163840
constexpr size_t WS_WPKC = WS_WPKF + 163840;                   // 3*2*18*32*16*2 = 110592
constexpr size_t WS_SA   = WS_WPKC + 110592;                   // 1024 f32
constexpr size_t WS_TA   = WS_SA   + 4096;
constexpr size_t WS_SF   = WS_TA   + 4096;                     // 64 f32
constexpr size_t WS_TF   = WS_SF   + 256;
constexpr size_t WS_POOL = WS_TF   + 256;                      // 4*64 f32
constexpr size_t WS_X5   = WS_POOL + 1024;                     // 4*256 f32

// ---------------------------------------------------------------------------
// BN fold: s = g*rsqrt(v+eps), t = b - m*s
__global__ void bnprep_kernel(const float* __restrict__ g, const float* __restrict__ bb,
                              const float* __restrict__ m, const float* __restrict__ v,
                              float* __restrict__ s, float* __restrict__ t, int n) {
    int i = blockIdx.x * blockDim.x + threadIdx.x;
    if (i >= n) return;
    float sc = g[i] * rsqrtf(v[i] + EPSBN);
    s[i] = sc;
    t[i] = bb[i] - m[i] * sc;
}

// ---------------------------------------------------------------------------
// Pack a dense f32 [O][CK] weight matrix into a 16-bit WMMA A-fragment layout (this code's k order):
// dst[tile][chunk][lane][16 f16], A 16x32 per wave, contiguous k (not the ISA table order):
//   dword v of lane l holds K = chunk*32 + (l&16 ? 16:0) + 2*v .. +1
__global__ void pack_kernel(const float* __restrict__ src, _Float16* __restrict__ dst,
                            int CK, int nchunks, int total) {
    int id = blockIdx.x * blockDim.x + threadIdx.x;
    if (id >= total) return;
    int per   = nchunks * 32;
    int tile  = id / per;
    int rm    = id - tile * per;
    int chunk = rm >> 5;
    int lane  = rm & 31;
    int o     = tile * 16 + (lane & 15);
    int hi    = (lane & 16) ? 16 : 0;
    _Float16* d = dst + (size_t)id * 16;
#pragma unroll
    for (int v = 0; v < 8; ++v) {
        int kk = chunk * 32 + hi + (v << 1);
        d[2 * v]     = (_Float16)src[(size_t)o * CK + kk];
        d[2 * v + 1] = (_Float16)src[(size_t)o * CK + kk + 1];
    }
}

// Pack the combined offset(18)+mask(9)+zero-pad(5) 32-row matrix for all 3
// branches: dst[r][tile(2)][chunk(18)][lane(32)][16]; rows 0..17 from ow3[r],
// rows 18..26 from mw3[r], rows 27..31 zero.
__global__ void pack_coords_kernel(const float* __restrict__ ow3,
                                   const float* __restrict__ mw3,
                                   _Float16* __restrict__ dst) {
    int id = blockIdx.x * blockDim.x + threadIdx.x;   // total 3*2*18*32
    if (id >= 3 * 2 * 18 * 32) return;
    int lane  = id & 31;
    int t2    = id >> 5;
    int chunk = t2 % 18;
    int t3    = t2 / 18;
    int tile  = t3 & 1;
    int r     = t3 >> 1;
    int o     = tile * 16 + (lane & 15);
    int hi    = (lane & 16) ? 16 : 0;
    _Float16* d = dst + (size_t)id * 16;
#pragma unroll
    for (int v = 0; v < 8; ++v) {
        int kk = chunk * 32 + hi + (v << 1);
#pragma unroll
        for (int e = 0; e < 2; ++e) {
            int k2 = kk + e;
            float val = 0.f;
            if (o < 18)      val = ow3[(size_t)r * 18 * 576 + (size_t)o * 576 + k2];
            else if (o < 27) val = mw3[(size_t)r * 9 * 576 + (size_t)(o - 18) * 576 + k2];
            d[2 * v + e] = (_Float16)val;
        }
    }
}

// ---------------------------------------------------------------------------
// Global average pool: one block per (c, b)
__global__ void __launch_bounds__(256) gpool_kernel(const float* __restrict__ x,
                                                    float* __restrict__ pool) {
    int c = blockIdx.x, b = blockIdx.y;
    const float* xp = x + ((size_t)b * Cin + c) * NPIX;
    float s = 0.f;
    for (int i = threadIdx.x; i < NPIX; i += 256) s += xp[i];
    __shared__ float red[256];
    red[threadIdx.x] = s;
    __syncthreads();
    for (int st = 128; st > 0; st >>= 1) {
        if (threadIdx.x < st) red[threadIdx.x] += red[threadIdx.x + st];
        __syncthreads();
    }
    if (threadIdx.x == 0) pool[b * Cin + c] = red[0] * (1.f / NPIX);
}

// x5[b][p] = relu(bn(gw @ pooled))
__global__ void gx5_kernel(const float* __restrict__ pool, const float* __restrict__ gw,
                           const float* __restrict__ gg, const float* __restrict__ gb,
                           const float* __restrict__ gm, const float* __restrict__ gv,
                           float* __restrict__ x5) {
    int b = blockIdx.x, p = threadIdx.x;
    float s = 0.f;
    for (int c = 0; c < Cin; ++c) s += gw[p * Cin + c] * pool[b * Cin + c];
    float sc = gg[p] * rsqrtf(gv[p] + EPSBN);
    float v  = (s - gm[p]) * sc + gb[p];
    x5[b * Pc + p] = v > 0.f ? v : 0.f;
}

// broadcast x5 into concat channels [1024..1280)
__global__ void gfill_kernel(const float* __restrict__ x5, _Float16* __restrict__ cat) {
    int b = blockIdx.y, pix = blockIdx.x, ch = threadIdx.x;
    cat[((size_t)b * NPIX + pix) * CATC + 1024 + ch] = (_Float16)x5[b * Pc + ch];
}

// ---------------------------------------------------------------------------
// WMMA offset/mask convs for the three 3x3 branches.
// One im2col LDS tile S[16 pixels][576] serves all 3 branches (it is branch-
// independent): 6 tile-waves (3 branches x 2 row-tiles of the padded 32-row
// [ow;mw;0] matrix) consume it; epilogue adds biases / base coords / sigmoid.
__global__ void __launch_bounds__(256)
coords3w_kernel(const float* __restrict__ x, const _Float16* __restrict__ wpkc,
                const float* __restrict__ ob3, const float* __restrict__ mb3,
                float* __restrict__ py, float* __restrict__ px, float* __restrict__ mk) {
    constexpr int CK = 576, RS_H = 584, RS_DW = 292, NCHUNK = 18;
    __shared__ alignas(16) _Float16 S[16 * RS_H];
    const int b = blockIdx.y, pix0 = blockIdx.x * 16, tid = threadIdx.x;
    const float* xb = x + (size_t)b * Cin * NPIX;

    for (int idx = tid; idx < 16 * CK; idx += 256) {
        int pl = idx / CK, ck = idx - pl * CK;
        int c = ck / 9, t = ck - c * 9;
        int pix = pix0 + pl, h = pix >> 7, w = pix & 127;
        int yy = h - 1 + t / 3, xx = w - 1 + t % 3;
        float v = (yy >= 0 && yy < Hh && xx >= 0 && xx < Ww)
                      ? xb[(size_t)c * NPIX + yy * Ww + xx] : 0.f;
        S[pl * RS_H + ck] = (_Float16)v;
    }
    __syncthreads();

    const int wv = tid >> 5, lane = tid & 31, n = lane & 15, hi = (lane & 16) ? 1 : 0;
    if (wv < 6) {                       // wave-uniform branch: EXEC all-1 inside
        int r = wv >> 1, tile = wv & 1;
        const v16h* av = (const v16h*)(wpkc + (size_t)(r * 2 + tile) * NCHUNK * 32 * 16);
        const unsigned* Sdw = (const unsigned*)S;
        v8f acc = {};
#pragma unroll
        for (int q = 0; q < NCHUNK; ++q) {
            v16h a = av[(size_t)q * 32 + lane];
            union { v16h h; unsigned u[8]; } bf;
            const unsigned* rowp = Sdw + n * RS_DW + q * 16 + hi * 8;
#pragma unroll
            for (int v = 0; v < 8; ++v) bf.u[v] = rowp[v];
            acc = __builtin_amdgcn_wmma_f32_16x16x32_f16(false, a, false, bf.h,
                                                         (short)0, acc, false, false);
        }
        int pix = pix0 + n, h = pix >> 7, w = pix & 127;
#pragma unroll
        for (int i = 0; i < 8; ++i) {
            int o = tile * 16 + i + hi * 8;          // padded channel 0..31
            float v = acc[i];
            if (o < 18) {
                int k = o >> 1;
                size_t oi = (((size_t)r * Bn + b) * 9 + k) * NPIX + pix;
                if ((o & 1) == 0) py[oi] = v + ob3[r * 18 + o] + (float)(h - 1 + k / 3);
                else              px[oi] = v + ob3[r * 18 + o] + (float)(w - 1 + k % 3);
            } else if (o < 27) {
                int k = o - 18;
                size_t oi = (((size_t)r * Bn + b) * 9 + k) * NPIX + pix;
                mk[oi] = 2.f / (1.f + __expf(-(v + mb3[r * 9 + k])));
            }
        }
    }
}

// 1x1 offset/mask convs (pad 0) — tiny (3x64 MACs/pixel), plain VALU is fine.
__global__ void __launch_bounds__(256)
coords1_kernel(const float* __restrict__ x,
               const float* __restrict__ ow1, const float* __restrict__ ob1,
               const float* __restrict__ mw1, const float* __restrict__ mb1,
               float* __restrict__ py, float* __restrict__ px, float* __restrict__ mk) {
    int b = blockIdx.y;
    int pix = blockIdx.x * 256 + threadIdx.x;
    int h = pix >> 7, w = pix & 127;
    const float* xb = x + (size_t)b * Cin * NPIX + pix;
    float a0 = 0.f, a1 = 0.f, a2 = 0.f;
    for (int c = 0; c < Cin; ++c) {
        float xv = xb[(size_t)c * NPIX];
        a0 += xv * ow1[c];
        a1 += xv * ow1[Cin + c];
        a2 += xv * mw1[c];
    }
    a0 += ob1[0]; a1 += ob1[1]; a2 += mb1[0];
    size_t o = (size_t)b * NPIX + pix;
    py[o] = a0 + (float)h;
    px[o] = a1 + (float)w;
    mk[o] = 2.f / (1.f + __expf(-a2));
}

// ---------------------------------------------------------------------------
// Deformable-conv GEMM: block = 16 pixels x 256 outputs, 16 waves.
// Phase A: per (pixel,tap) corner offsets + mask-folded bilinear weights -> LDS
// Phase B: sampled S[16][64*KD] f16 -> LDS (2x ds_load_b128 + 4 gathers + 4 FMA)
// Phase C: per-wave 16x16 tile, 2*KD chained WMMAs, fused BN+ReLU, f16 store.
template <int KD>
__global__ void __launch_bounds__(512)
deform_wmma(const float* __restrict__ x,
            const float* __restrict__ py, const float* __restrict__ px,
            const float* __restrict__ mk,
            const _Float16* __restrict__ wpack,
            const float* __restrict__ sA, const float* __restrict__ tA,
            _Float16* __restrict__ cat, int cbase) {
    constexpr int CK     = Cin * KD;
    constexpr int NCHUNK = CK / 32;
    constexpr int RS_H   = CK + 8;         // padded row stride (f16) vs LDS banks
    constexpr int RS_DW  = RS_H >> 1;
    __shared__ alignas(16) _Float16 S[16 * RS_H];
    __shared__ alignas(16) int   cofs[16 * KD * 4];
    __shared__ alignas(16) float cwts[16 * KD * 4];

    const int b    = blockIdx.y;
    const int pix0 = blockIdx.x * 16;
    const int tid  = threadIdx.x;
    const float* xb = x + (size_t)b * Cin * NPIX;

    // Phase A: bilinear corner data, once per (pixel, tap)
    for (int idx = tid; idx < 16 * KD; idx += 512) {
        int pl = idx / KD, k = idx - pl * KD;
        size_t ci = ((size_t)b * KD + k) * NPIX + (pix0 + pl);
        float pyv = py[ci], pxv = px[ci], m = mk[ci];
        float y0f = floorf(pyv), x0f = floorf(pxv);
        int y0 = (int)y0f, x0 = (int)x0f;
        float wy = pyv - y0f, wx = pxv - x0f;
        float w00 = (1.f - wy) * (1.f - wx) * m, w01 = (1.f - wy) * wx * m;
        float w10 = wy * (1.f - wx) * m,        w11 = wy * wx * m;
        int y0c = min(max(y0, 0), Hh - 1), y1c = min(max(y0 + 1, 0), Hh - 1);
        int x0c = min(max(x0, 0), Ww - 1), x1c = min(max(x0 + 1, 0), Ww - 1);
        bool y0v = (y0 >= 0) && (y0 < Hh), y1v = (y0 + 1 >= 0) && (y0 + 1 < Hh);
        bool x0v = (x0 >= 0) && (x0 < Ww), x1v = (x0 + 1 >= 0) && (x0 + 1 < Ww);
        cofs[idx * 4 + 0] = y0c * Ww + x0c;  cwts[idx * 4 + 0] = (y0v && x0v) ? w00 : 0.f;
        cofs[idx * 4 + 1] = y0c * Ww + x1c;  cwts[idx * 4 + 1] = (y0v && x1v) ? w01 : 0.f;
        cofs[idx * 4 + 2] = y1c * Ww + x0c;  cwts[idx * 4 + 2] = (y1v && x0v) ? w10 : 0.f;
        cofs[idx * 4 + 3] = y1c * Ww + x1c;  cwts[idx * 4 + 3] = (y1v && x1v) ? w11 : 0.f;
    }
    __syncthreads();

    // Phase B: gather + weight into the sampled operand
    for (int idx = tid; idx < 16 * CK; idx += 512) {
        int pl = idx / CK, ck = idx - pl * CK;
        int c = ck / KD, k = ck - c * KD;
        int pr = (pl * KD + k) * 4;
        const float* xc = xb + (size_t)c * NPIX;
        float v = xc[cofs[pr + 0]] * cwts[pr + 0] + xc[cofs[pr + 1]] * cwts[pr + 1]
                + xc[cofs[pr + 2]] * cwts[pr + 2] + xc[cofs[pr + 3]] * cwts[pr + 3];
        S[pl * RS_H + ck] = (_Float16)v;
    }
    __syncthreads();

    // Phase C: WMMA
    const int wv   = tid >> 5;
    const int lane = tid & 31;
    const int n    = lane & 15;
    const int hi   = (lane & 16) ? 1 : 0;
    const unsigned* Sdw = (const unsigned*)S;
    const v16h* wfa = (const v16h*)wpack;

    v8f acc = {};
#pragma unroll
    for (int q = 0; q < NCHUNK; ++q) {
        v16h a = wfa[(size_t)(wv * NCHUNK + q) * 32 + lane];
        union { v16h h; unsigned u[8]; } bf;
        const unsigned* rowp = Sdw + n * RS_DW + q * 16 + hi * 8;
#pragma unroll
        for (int v = 0; v < 8; ++v) bf.u[v] = rowp[v];
        acc = __builtin_amdgcn_wmma_f32_16x16x32_f16(false, a, false, bf.h,
                                                     (short)0, acc, false, false);
    }

    size_t ob = ((size_t)b * NPIX + pix0 + n) * CATC + cbase;
#pragma unroll
    for (int i = 0; i < 8; ++i) {
        int ol = wv * 16 + i + hi * 8;       // C layout: VGPR i -> M = i (+8 hi half)
        float v = acc[i] * sA[ol] + tA[ol];
        v = v > 0.f ? v : 0.f;
        cat[ob + ol] = (_Float16)v;
    }
}

// ---------------------------------------------------------------------------
// 1x1 fusion conv over the 1280-channel concat, fused BN+ReLU, f32 out.
__global__ void __launch_bounds__(128)
fuse_wmma(const _Float16* __restrict__ cat, const _Float16* __restrict__ fwp,
          const float* __restrict__ sF, const float* __restrict__ tF,
          float* __restrict__ out) {
    constexpr int RS_DW = (CATC + 8) >> 1;  // 644 dwords / pixel row
    __shared__ alignas(16) _Float16 T[16 * (CATC + 8)];
    const int b = blockIdx.y, pix0 = blockIdx.x * 16, tid = threadIdx.x;
    unsigned* Tdw = (unsigned*)T;
    const unsigned* catdw = (const unsigned*)cat;
    for (int i = tid; i < 16 * (CATC / 2); i += 128) {
        int pl = i / (CATC / 2), d = i - pl * (CATC / 2);
        Tdw[pl * RS_DW + d] = catdw[((size_t)b * NPIX + pix0 + pl) * (CATC / 2) + d];
    }
    __syncthreads();

    const int wv = tid >> 5, lane = tid & 31, n = lane & 15, hi = (lane & 16) ? 1 : 0;
    const v16h* av = (const v16h*)fwp;
    v8f acc = {};
    constexpr int NCH = CATC / 32;          // 40
#pragma unroll 8
    for (int q = 0; q < NCH; ++q) {
        v16h a = av[(size_t)(wv * NCH + q) * 32 + lane];
        union { v16h h; unsigned u[8]; } bf;
        const unsigned* rowp = (const unsigned*)T + n * RS_DW + q * 16 + hi * 8;
#pragma unroll
        for (int v = 0; v < 8; ++v) bf.u[v] = rowp[v];
        acc = __builtin_amdgcn_wmma_f32_16x16x32_f16(false, a, false, bf.h,
                                                     (short)0, acc, false, false);
    }
#pragma unroll
    for (int i = 0; i < 8; ++i) {
        int o = wv * 16 + i + hi * 8;        // o in [0,64)
        float v = acc[i] * sF[o] + tF[o];
        v = v > 0.f ? v : 0.f;
        out[(((size_t)b * 64 + o) << 14) + pix0 + n] = v;
    }
}

// ---------------------------------------------------------------------------
extern "C" void kernel_launch(void* const* d_in, const int* in_sizes, int n_in,
                              void* d_out, int out_size, void* d_ws, size_t ws_size,
                              hipStream_t stream) {
    const float* x   = (const float*)d_in[0];
    const float* ow1 = (const float*)d_in[1];
    const float* ob1 = (const float*)d_in[2];
    const float* mw1 = (const float*)d_in[3];
    const float* mb1 = (const float*)d_in[4];
    const float* rw1 = (const float*)d_in[5];
    const float* g1  = (const float*)d_in[6];
    const float* b1  = (const float*)d_in[7];
    const float* m1  = (const float*)d_in[8];
    const float* v1  = (const float*)d_in[9];
    const float* ow3 = (const float*)d_in[10];
    const float* ob3 = (const float*)d_in[11];
    const float* mw3 = (const float*)d_in[12];
    const float* mb3 = (const float*)d_in[13];
    const float* rw3 = (const float*)d_in[14];
    const float* g3  = (const float*)d_in[15];
    const float* b3  = (const float*)d_in[16];
    const float* m3  = (const float*)d_in[17];
    const float* v3  = (const float*)d_in[18];
    const float* gw  = (const float*)d_in[19];
    const float* gg  = (const float*)d_in[20];
    const float* gbb = (const float*)d_in[21];
    const float* gm  = (const float*)d_in[22];
    const float* gv  = (const float*)d_in[23];
    const float* fw  = (const float*)d_in[24];
    const float* fg  = (const float*)d_in[25];
    const float* fb  = (const float*)d_in[26];
    const float* fm  = (const float*)d_in[27];
    const float* fv  = (const float*)d_in[28];

    char* ws = (char*)d_ws;
    _Float16* cat  = (_Float16*)(ws + WS_CAT);
    float* py3 = (float*)(ws + WS_PY3);
    float* px3 = (float*)(ws + WS_PX3);
    float* mk3 = (float*)(ws + WS_MK3);
    float* py1 = (float*)(ws + WS_PY1);
    float* px1 = (float*)(ws + WS_PX1);
    float* mk1 = (float*)(ws + WS_MK1);
    _Float16* wpk1 = (_Float16*)(ws + WS_WPK1);
    _Float16* wpk3 = (_Float16*)(ws + WS_WPK3);
    _Float16* wpkf = (_Float16*)(ws + WS_WPKF);
    _Float16* wpkc = (_Float16*)(ws + WS_WPKC);
    float* sA   = (float*)(ws + WS_SA);
    float* tA   = (float*)(ws + WS_TA);
    float* sF   = (float*)(ws + WS_SF);
    float* tF   = (float*)(ws + WS_TF);
    float* pool = (float*)(ws + WS_POOL);
    float* x5   = (float*)(ws + WS_X5);

    // BN folds (branch 0 = 1x1, branches 1..3 = 3x3, fusion)
    bnprep_kernel<<<1, 256, 0, stream>>>(g1, b1, m1, v1, sA, tA, Pc);
    for (int r = 0; r < 3; ++r)
        bnprep_kernel<<<1, 256, 0, stream>>>(g3 + r * Pc, b3 + r * Pc, m3 + r * Pc,
                                             v3 + r * Pc, sA + (r + 1) * Pc,
                                             tA + (r + 1) * Pc, Pc);
    bnprep_kernel<<<1, 64, 0, stream>>>(fg, fb, fm, fv, sF, tF, 64);

    // Weight packing into WMMA A-fragment layout (sources are [O][CK] contiguous)
    pack_kernel<<<(16 * 2 * 32 + 255) / 256, 256, 0, stream>>>(rw1, wpk1, 64, 2, 16 * 2 * 32);
    pack_kernel<<<(48 * 18 * 32 + 255) / 256, 256, 0, stream>>>(rw3, wpk3, 576, 18, 48 * 18 * 32);
    pack_kernel<<<(4 * 40 * 32 + 255) / 256, 256, 0, stream>>>(fw, wpkf, 1280, 40, 4 * 40 * 32);
    pack_coords_kernel<<<(3 * 2 * 18 * 32 + 255) / 256, 256, 0, stream>>>(ow3, mw3, wpkc);

    // Global-pool branch -> concat channels [1024,1280)
    gpool_kernel<<<dim3(Cin, Bn), 256, 0, stream>>>(x, pool);
    gx5_kernel<<<Bn, Pc, 0, stream>>>(pool, gw, gg, gbb, gm, gv, x5);
    gfill_kernel<<<dim3(NPIX, Bn), Pc, 0, stream>>>(x5, cat);

    // Sample coordinates + masks (3x3 branches via WMMA, 1x1 via VALU)
    coords1_kernel<<<dim3(NPIX / 256, Bn), 256, 0, stream>>>(x, ow1, ob1, mw1, mb1,
                                                             py1, px1, mk1);
    coords3w_kernel<<<dim3(NPIX / 16, Bn), 256, 0, stream>>>(x, wpkc, ob3, mb3,
                                                             py3, px3, mk3);

    // Deformable-conv GEMMs (WMMA), outputs -> concat channels
    deform_wmma<1><<<dim3(NPIX / 16, Bn), 512, 0, stream>>>(x, py1, px1, mk1, wpk1,
                                                            sA, tA, cat, 0);
    for (int r = 0; r < 3; ++r) {
        size_t co = (size_t)r * Bn * 9 * NPIX;
        deform_wmma<9><<<dim3(NPIX / 16, Bn), 512, 0, stream>>>(
            x, py3 + co, px3 + co, mk3 + co,
            wpk3 + (size_t)r * 16 * 18 * 32 * 16,
            sA + (r + 1) * Pc, tA + (r + 1) * Pc, cat, (r + 1) * Pc);
    }

    // Fusion 1x1 conv (WMMA) + BN + ReLU -> d_out
    fuse_wmma<<<dim3(NPIX / 16, Bn), 128, 0, stream>>>(cat, wpkf, sF, tF, (float*)d_out);
}
